// GCNEncoder_20804821582421
// MI455X (gfx1250) — compile-verified, matrix operand order checked by reading, not run on hardware
//
#include <hip/hip_runtime.h>

// GCN 2-layer forward for MI455X (gfx1250, wave32).
// Dense GEMMs via v_wmma_f32_16x16x32_bf16 with bf16x3 split (≈fp32 accuracy);
// edge aggregation via native global_atomic_add_f32 (L2-resident scatter-add).

#define N_NODES 100000
#define IN_DIM  128
#define HID_DIM 128
#define OUT_DIM 64

typedef __attribute__((ext_vector_type(16))) __bf16 v16bf;
typedef __attribute__((ext_vector_type(8)))  float  v8f;

union V16BF { v16bf v; unsigned short s[16]; };

__device__ __forceinline__ unsigned short f32_to_bf16_rne(float f) {
  union { float f; unsigned u; } c; c.f = f;
  unsigned u = c.u;
  unsigned r = u + 0x7FFFu + ((u >> 16) & 1u);  // round-to-nearest-even
  return (unsigned short)(r >> 16);
}
__device__ __forceinline__ float bf16_to_f32(unsigned short h) {
  union { unsigned u; float f; } c; c.u = ((unsigned)h) << 16;
  return c.f;
}

__device__ __forceinline__ void atomic_add_f32(float* p, float v) {
  // non-returning fp32 atomic add; device scope, default temporal hint
  asm volatile("global_atomic_add_f32 %0, %1, off" :: "v"(p), "v"(v) : "memory");
}

// ---------------------------------------------------------------- degree/norm
__global__ void k_init_deg(float* __restrict__ deg, int n) {
  int i = blockIdx.x * blockDim.x + threadIdx.x;
  if (i < n) deg[i] = 1.0f;  // self-loop contributes 1
}

__global__ void k_degree_count(const int* __restrict__ dst, float* __restrict__ deg, int nE) {
  int i = blockIdx.x * blockDim.x + threadIdx.x;
  if (i < nE) atomic_add_f32(&deg[dst[i]], 1.0f);
}

__global__ void k_dinv(float* __restrict__ deg, int n) {
  int i = blockIdx.x * blockDim.x + threadIdx.x;
  if (i < n) deg[i] = rsqrtf(deg[i]);  // deg >= 1 always
}

// ------------------------------------------------------------------ WMMA GEMM
// Y[M x NCOLS] = (RELU? max(X,0):X)[M x KDIM] @ W[KDIM x NCOLS]
// One wave per 16x16 tile; blockDim = 32 * (NCOLS/16); gridDim.x = M/16.
// A layout (16-bit, 16x32): lane l holds row (l&15); k-half = (l>>4)*8;
// elements e<8 -> K = khalf+e, e>=8 -> K = 16+khalf+(e-8).
// B layout mirrors A with column index (l&15).
template<int KDIM, int NCOLS, bool RELU>
__global__ void k_gemm_bf16x3_wmma(const float* __restrict__ X,
                                   const float* __restrict__ W,
                                   float* __restrict__ Y) {
  const int lane  = threadIdx.x & 31;
  const int ntile = threadIdx.x >> 5;
  const int mtile = blockIdx.x;
  const int lrow  = lane & 15;
  const int khalf = (lane >> 4) << 3;

  const int row  = mtile * 16 + lrow;   // A row this lane carries
  const int ncol = ntile * 16 + lrow;   // B column this lane carries

  v8f acc = {0.f, 0.f, 0.f, 0.f, 0.f, 0.f, 0.f, 0.f};

  for (int k0 = 0; k0 < KDIM; k0 += 32) {
    float a[16], b[16];
    const float* ap = X + (size_t)row * KDIM + k0 + khalf;
#pragma unroll
    for (int e = 0; e < 8; ++e) a[e] = ap[e];          // contiguous 32B x2
#pragma unroll
    for (int e = 0; e < 8; ++e) a[8 + e] = ap[16 + e];
    if (RELU) {
#pragma unroll
      for (int e = 0; e < 16; ++e) a[e] = fmaxf(a[e], 0.0f);
    }
    const float* bp = W + (size_t)(k0 + khalf) * NCOLS + ncol;
#pragma unroll
    for (int e = 0; e < 8; ++e) b[e] = bp[(size_t)e * NCOLS];
#pragma unroll
    for (int e = 0; e < 8; ++e) b[8 + e] = bp[(size_t)(16 + e) * NCOLS];

    V16BF ahi, alo, bhi, blo;
#pragma unroll
    for (int e = 0; e < 16; ++e) {
      unsigned short h = f32_to_bf16_rne(a[e]);
      ahi.s[e] = h;
      alo.s[e] = f32_to_bf16_rne(a[e] - bf16_to_f32(h));
      unsigned short g = f32_to_bf16_rne(b[e]);
      bhi.s[e] = g;
      blo.s[e] = f32_to_bf16_rne(b[e] - bf16_to_f32(g));
    }
    // bf16x3: hi*hi + lo*hi + hi*lo, f32 accumulate
    acc = __builtin_amdgcn_wmma_f32_16x16x32_bf16(false, ahi.v, false, bhi.v, (short)0, acc, false, false);
    acc = __builtin_amdgcn_wmma_f32_16x16x32_bf16(false, alo.v, false, bhi.v, (short)0, acc, false, false);
    acc = __builtin_amdgcn_wmma_f32_16x16x32_bf16(false, ahi.v, false, blo.v, (short)0, acc, false, false);
  }

  // C/D layout: VGPR j, lanes 0-15 -> M=j, lanes 16-31 -> M=j+8; N = lane&15
  float* yp = Y + (size_t)(mtile * 16 + ((lane >> 4) << 3)) * NCOLS + ntile * 16 + lrow;
#pragma unroll
  for (int j = 0; j < 8; ++j) yp[(size_t)j * NCOLS] = acc[j];
}

// ------------------------------------------------- self-loop + bias init pass
// Out[i][c] = bias[c] + T[i][c] * dinv[i]^2   (one float4 per thread)
template<int DIM>
__global__ void k_init_agg(const float* __restrict__ T, const float* __restrict__ bias,
                           const float* __restrict__ dinv, float* __restrict__ Out,
                           int nNodes) {
  int i = blockIdx.x * blockDim.x + threadIdx.x;
  int node = i / (DIM / 4);
  int c    = (i % (DIM / 4)) * 4;
  if (node >= nNodes) return;
  float sw = dinv[node]; sw *= sw;
  const float4 t = *(const float4*)(T + (size_t)node * DIM + c);
  const float4 bb = *(const float4*)(bias + c);
  float4 o;
  o.x = bb.x + t.x * sw;  o.y = bb.y + t.y * sw;
  o.z = bb.z + t.z * sw;  o.w = bb.w + t.w * sw;
  *(float4*)(Out + (size_t)node * DIM + c) = o;
}

// ----------------------------------------------------------- edge scatter-add
// One wave per edge; lane covers DIM/32 consecutive floats -> coalesced 512B/256B
template<int DIM>
__global__ void k_scatter_edges(const float* __restrict__ T, const int* __restrict__ src,
                                const int* __restrict__ dst, const float* __restrict__ dinv,
                                float* __restrict__ Out, int nE) {
  int gid  = blockIdx.x * blockDim.x + threadIdx.x;
  int edge = gid >> 5;
  int lane = gid & 31;
  if (edge >= nE) return;
  int s = src[edge], d = dst[edge];
  float nrm = dinv[s] * dinv[d];
  constexpr int PER = DIM / 32;
  const float* tp = T + (size_t)s * DIM + lane * PER;
  float*       op = Out + (size_t)d * DIM + lane * PER;
#pragma unroll
  for (int j = 0; j < PER; ++j) atomic_add_f32(op + j, tp[j] * nrm);
}

// ---------------------------------------------------------------------- entry
extern "C" void kernel_launch(void* const* d_in, const int* in_sizes, int n_in,
                              void* d_out, int out_size, void* d_ws, size_t ws_size,
                              hipStream_t stream) {
  const float* x  = (const float*)d_in[0];
  const int*   ei = (const int*)d_in[1];
  const float* W1 = (const float*)d_in[2];
  const float* b1 = (const float*)d_in[3];
  const float* W2 = (const float*)d_in[4];
  const float* b2 = (const float*)d_in[5];
  float* out = (float*)d_out;

  const int nE  = in_sizes[1] / 2;
  const int* src = ei;        // edge_index[0]
  const int* dst = ei + nE;   // edge_index[1]

  // workspace layout (floats): dinv[N] | tA[N*128] | hB[N*128]  (~102.8 MB)
  float* ws   = (float*)d_ws;
  float* dinv = ws;
  float* tA   = ws + 100096;                       // 128-float aligned
  float* hB   = tA + (size_t)N_NODES * HID_DIM;

  const int T = 256;
  const int gN = (N_NODES + T - 1) / T;
  const int gE = (nE + T - 1) / T;
  const int gEdgeWave = (int)(((size_t)nE * 32 + T - 1) / T);
  const int mTiles = N_NODES / 16;  // 6250 exactly

  // --- normalization coefficients ---
  k_init_deg<<<gN, T, 0, stream>>>(dinv, N_NODES);
  k_degree_count<<<gE, T, 0, stream>>>(dst, dinv, nE);
  k_dinv<<<gN, T, 0, stream>>>(dinv, N_NODES);

  // --- layer 1: t1 = x@W1; hB = b1 + self + scatter(t1) ---
  k_gemm_bf16x3_wmma<IN_DIM, HID_DIM, false>
      <<<mTiles, 32 * (HID_DIM / 16), 0, stream>>>(x, W1, tA);
  k_init_agg<HID_DIM>
      <<<(int)(((size_t)N_NODES * (HID_DIM / 4) + T - 1) / T), T, 0, stream>>>(tA, b1, dinv, hB, N_NODES);
  k_scatter_edges<HID_DIM><<<gEdgeWave, T, 0, stream>>>(tA, src, dst, dinv, hB, nE);

  // --- layer 2: t2 = relu(hB)@W2 (relu fused in A-load); out = b2 + self + scatter ---
  k_gemm_bf16x3_wmma<HID_DIM, OUT_DIM, true>
      <<<mTiles, 32 * (OUT_DIM / 16), 0, stream>>>(hB, W2, tA);
  k_init_agg<OUT_DIM>
      <<<(int)(((size_t)N_NODES * (OUT_DIM / 4) + T - 1) / T), T, 0, stream>>>(tA, b2, dinv, out, N_NODES);
  k_scatter_edges<OUT_DIM><<<gEdgeWave, T, 0, stream>>>(tA, src, dst, dinv, out, nE);
}
